// SUBNET_15290083574166
// MI455X (gfx1250) — matrix operand order repaired and checked by reading, not compiled, not run on hardware
//
#include <hip/hip_runtime.h>
#include <hip/hip_bf16.h>

// ---------------------------------------------------------------------------
// SUBNET simulation for MI455X (gfx1250), bf16 WMMA with fp32 accumulation.
//
//   1) 12x swizzle kernels: f32 weights -> bf16 WMMA-B-fragment layout.
//   2) encoder kernel: x0 = MLP_res(concat(upast,ypast))        [B x 128]
//   3) scan kernel: 128 blocks x 32 batch rows; for t in 0..255:
//        y[:,t] = MLP_res_h(x);  x = MLP_res_f([x|u_t])
//      state lives in LDS for the whole scan; 2x2 register blocking so each
//      B fragment feeds 2 wmma and each A fragment feeds 2 wmma (halves both
//      L2 weight traffic and LDS activation traffic vs 1x1 blocking).
// ---------------------------------------------------------------------------

typedef __attribute__((ext_vector_type(16))) __bf16 v16bf;
typedef __attribute__((ext_vector_type(8)))  __bf16 v8bf;
typedef __attribute__((ext_vector_type(8)))  float  v8f;

#define B_TOT 4096
#define T_LEN 256
#define NX    128
#define NU    16
#define NY    16
#define HDIM  256
#define BM    32      // batch rows per block (2 M-tiles)

static __device__ inline __bf16 f2bf(float f) {
  // round-to-nearest-even f32 -> bf16 via integer ops
  unsigned int u = __float_as_uint(f);
  unsigned int r = (u + 0x7FFFu + ((u >> 16) & 1u)) >> 16;
  unsigned short s = (unsigned short)r;
  return *reinterpret_cast<__bf16*>(&s);
}

static __device__ inline v8f wmma_bf16(v16bf a, v16bf b, v8f c) {
  return __builtin_amdgcn_wmma_f32_16x16x32_bf16(
      /*neg_a=*/false, a, /*neg_b=*/false, b,
      /*c_mod=*/(short)0, c, /*reuse_a=*/false, /*reuse_b=*/false);
}

// A-fragment (16x32 bf16, row-major in LDS, row stride `stride` elems).
// ISA layout: lane L (r=L&15, h=L>>4) element e: K = (e>>3)*16 + h*8 + (e&7)
// -> two contiguous 8-elem (16B) chunks at K = h*8 and K = 16 + h*8.
static __device__ inline v16bf load_a_frag(const __bf16* base, int stride, int lane) {
  const int r = lane & 15;
  const int h = lane >> 4;
  const __bf16* row = base + r * stride;
  v8bf lo = *(const v8bf*)(row + h * 8);
  v8bf hi = *(const v8bf*)(row + 16 + h * 8);
  v16bf a;
  ((v8bf*)&a)[0] = lo;
  ((v8bf*)&a)[1] = hi;
  return a;
}

// B-fragment: pre-swizzled in global memory, lane's 16 elems contiguous (32B).
static __device__ inline v16bf load_b_frag(const __bf16* gtile, int lane) {
  return *(const v16bf*)(gtile + lane * 16);
}

// 2x2 tile block: acc[mt][j] += A(mt) * B(nt0+j), K = KT*32.
// B layout: tile (kt,nt) at ((size_t)(nt*KT + kt)) * 512 elems.
static __device__ inline void gemm_block22(const __bf16* ldsA, int strideA,
                                           const __bf16* gB, int KT, int nt0,
                                           int lane, v8f& c00, v8f& c01,
                                           v8f& c10, v8f& c11) {
  const __bf16* bt0 = gB + (size_t)(nt0 * KT) * 512;
  const __bf16* bt1 = gB + (size_t)((nt0 + 1) * KT) * 512;
  for (int kt = 0; kt < KT; ++kt) {
    v16bf A0 = load_a_frag(ldsA + kt * 32, strideA, lane);
    v16bf A1 = load_a_frag(ldsA + 16 * strideA + kt * 32, strideA, lane);
    v16bf B0 = load_b_frag(bt0 + (size_t)kt * 512, lane);
    v16bf B1 = load_b_frag(bt1 + (size_t)kt * 512, lane);
    c00 = wmma_bf16(A0, B0, c00);
    c10 = wmma_bf16(A1, B0, c10);
    c01 = wmma_bf16(A0, B1, c01);
    c11 = wmma_bf16(A1, B1, c11);
  }
}

// 2x1 tile block: acc[mt] += A(mt) * B(nt).
static __device__ inline void gemm_block21(const __bf16* ldsA, int strideA,
                                           const __bf16* gB, int KT, int nt,
                                           int lane, v8f& c0, v8f& c1) {
  const __bf16* bt = gB + (size_t)(nt * KT) * 512;
  for (int kt = 0; kt < KT; ++kt) {
    v16bf A0 = load_a_frag(ldsA + kt * 32, strideA, lane);
    v16bf A1 = load_a_frag(ldsA + 16 * strideA + kt * 32, strideA, lane);
    v16bf B0 = load_b_frag(bt + (size_t)kt * 512, lane);
    c0 = wmma_bf16(A0, B0, c0);
    c1 = wmma_bf16(A1, B0, c1);
  }
}

// Store C/D tile (fp32 acc) to LDS bf16 with bias and optional tanh.
// C/D layout: lane L (n=L&15, h=L>>4), element v -> row v + h*8, col n.
static __device__ inline void store_act(v8f acc, __bf16* lds, int stride,
                                        int mt, int nt,
                                        const float* __restrict__ bias,
                                        int lane, bool do_tanh) {
  const int n = lane & 15;
  const int h = lane >> 4;
  const float bv = bias[nt * 16 + n];
#pragma unroll
  for (int v = 0; v < 8; ++v) {
    float val = acc[v] + bv;
    if (do_tanh) val = tanhf(val);
    lds[(mt * 16 + v + h * 8) * stride + nt * 16 + n] = f2bf(val);
  }
}

// ---------------------------------------------------------------------------
// Weight swizzle: f32 [Kreal x N] row-major -> bf16 WMMA-B fragments,
// K padded to Kpad (multiple of 32) with zeros.
// ---------------------------------------------------------------------------
__global__ void swizzle_weight_kernel(const float* __restrict__ W,
                                      __bf16* __restrict__ outw,
                                      int Kreal, int Kpad, int N) {
  int idx = blockIdx.x * blockDim.x + threadIdx.x;
  int total = Kpad * N;
  if (idx >= total) return;
  int KT = Kpad >> 5;
  int tile = idx >> 9;        // 512 elems per 32x16 tile
  int within = idx & 511;
  int lane = within >> 4;
  int e = within & 15;
  int nt = tile / KT;
  int kt = tile - nt * KT;
  int K = kt * 32 + (e >> 3) * 16 + (lane >> 4) * 8 + (e & 7);
  int n = nt * 16 + (lane & 15);
  float v = (K < Kreal) ? W[(size_t)K * N + n] : 0.0f;
  outw[idx] = f2bf(v);
}

// ---------------------------------------------------------------------------
// Encoder: x0 = MLP_res(z), z = [upast|ypast] flattened (K=1024).
// 32 batch rows per block, 8 waves; hid stages: 32 tiles -> 2x2 per wave.
// ---------------------------------------------------------------------------
__global__ __launch_bounds__(256) void encoder_kernel(
    const float* __restrict__ upast, const float* __restrict__ ypast,
    const __bf16* __restrict__ wrW, const __bf16* __restrict__ wW1,
    const __bf16* __restrict__ wW2, const __bf16* __restrict__ wW3,
    const float* __restrict__ rb, const float* __restrict__ b1,
    const float* __restrict__ b2, const float* __restrict__ b3,
    __bf16* __restrict__ x0) {
  __shared__ __align__(32) __bf16 sZ[BM * 1024];
  __shared__ __align__(32) __bf16 sH1[BM * HDIM];
  __shared__ __align__(32) __bf16 sH2[BM * HDIM];
  const int tid = threadIdx.x;
  const int lane = tid & 31;
  const int w = tid >> 5;
  const int row0 = blockIdx.x * BM;

  for (int idx = tid; idx < BM * 1024; idx += 256) {
    int r = idx >> 10, c = idx & 1023;
    float v = (c < 512) ? upast[(size_t)(row0 + r) * 512 + c]
                        : ypast[(size_t)(row0 + r) * 512 + (c - 512)];
    sZ[idx] = f2bf(v);
  }
  __syncthreads();

  // hid1 = tanh(z @ W1 + b1)   K=1024 (KT=32), N=256: 2x2 block per wave
  {
    v8f c00 = {}, c01 = {}, c10 = {}, c11 = {};
    gemm_block22(sZ, 1024, wW1, 32, 2 * w, lane, c00, c01, c10, c11);
    store_act(c00, sH1, HDIM, 0, 2 * w,     b1, lane, true);
    store_act(c01, sH1, HDIM, 0, 2 * w + 1, b1, lane, true);
    store_act(c10, sH1, HDIM, 1, 2 * w,     b1, lane, true);
    store_act(c11, sH1, HDIM, 1, 2 * w + 1, b1, lane, true);
  }
  __syncthreads();

  // hid2 = tanh(hid1 @ W2 + b2)   KT=8
  {
    v8f c00 = {}, c01 = {}, c10 = {}, c11 = {};
    gemm_block22(sH1, HDIM, wW2, 8, 2 * w, lane, c00, c01, c10, c11);
    store_act(c00, sH2, HDIM, 0, 2 * w,     b2, lane, true);
    store_act(c01, sH2, HDIM, 0, 2 * w + 1, b2, lane, true);
    store_act(c10, sH2, HDIM, 1, 2 * w,     b2, lane, true);
    store_act(c11, sH2, HDIM, 1, 2 * w + 1, b2, lane, true);
  }
  __syncthreads();

  // x0 = z @ rW + hid2 @ W3 + rb + b3   N=128: nt = w, 2x1 block
  {
    v8f c0 = {}, c1 = {};
    gemm_block21(sZ, 1024, wrW, 32, w, lane, c0, c1);
    gemm_block21(sH2, HDIM, wW3, 8, w, lane, c0, c1);
    int n = lane & 15, h = lane >> 4;
    float bb = rb[w * 16 + n] + b3[w * 16 + n];
#pragma unroll
    for (int v = 0; v < 8; ++v) {
      x0[(size_t)(row0 + v + h * 8) * NX + w * 16 + n] = f2bf(c0[v] + bb);
      x0[(size_t)(row0 + 16 + v + h * 8) * NX + w * 16 + n] = f2bf(c1[v] + bb);
    }
  }
}

// ---------------------------------------------------------------------------
// Scan: each block owns BM=32 batch rows in LDS for all T=256 steps.
// Per step: y_t = h(x_t) (pre-update state, matches reference), then
// x_{t+1} = f([x_t | u_t]) (K=144 zero-padded to 160).
// ---------------------------------------------------------------------------
__global__ __launch_bounds__(256) void scan_kernel(
    const __bf16* __restrict__ x0, const float* __restrict__ ufuture,
    const __bf16* __restrict__ frW, const __bf16* __restrict__ fW1,
    const __bf16* __restrict__ fW2, const __bf16* __restrict__ fW3,
    const float* __restrict__ f_rb, const float* __restrict__ f_b1,
    const float* __restrict__ f_b2, const float* __restrict__ f_b3,
    const __bf16* __restrict__ hrW, const __bf16* __restrict__ hW1,
    const __bf16* __restrict__ hW2, const __bf16* __restrict__ hW3,
    const float* __restrict__ h_rb, const float* __restrict__ h_b1,
    const float* __restrict__ h_b2, const float* __restrict__ h_b3,
    float* __restrict__ out) {
  __shared__ __align__(32) __bf16 sX[BM * 160];   // [x(128) | u(16) | pad(16)]
  __shared__ __align__(32) __bf16 sH1[BM * HDIM];
  __shared__ __align__(32) __bf16 sH2[BM * HDIM];
  const int tid = threadIdx.x;
  const int lane = tid & 31;
  const int w = tid >> 5;
  const int row0 = blockIdx.x * BM;

  for (int idx = tid; idx < BM * NX; idx += 256) {
    int r = idx >> 7, c = idx & 127;
    sX[r * 160 + c] = x0[(size_t)(row0 + r) * NX + c];
  }
  for (int idx = tid; idx < BM * 16; idx += 256) {
    int r = idx >> 4, c = idx & 15;
    sX[r * 160 + 144 + c] = f2bf(0.0f);           // K-pad stays zero
  }
  __syncthreads();

  for (int t = 0; t < T_LEN; ++t) {
    { // load u_t (32x16, two elements per thread)
      for (int idx = tid; idx < BM * NU; idx += 256) {
        int r = idx >> 4, c = idx & 15;
        sX[r * 160 + 128 + c] =
            f2bf(ufuture[((size_t)(row0 + r) * T_LEN + t) * NU + c]);
      }
      if (t + 1 < T_LEN && tid < BM) { // prefetch next step's u slab
        __builtin_prefetch(
            &ufuture[((size_t)(row0 + tid) * T_LEN + t + 1) * NU], 0, 3);
      }
    }
    __syncthreads();

    // ---- h network on current x (pre-update state) ----
    {                                            // hid1h: KT=4
      v8f c00 = {}, c01 = {}, c10 = {}, c11 = {};
      gemm_block22(sX, 160, hW1, 4, 2 * w, lane, c00, c01, c10, c11);
      store_act(c00, sH1, HDIM, 0, 2 * w,     h_b1, lane, true);
      store_act(c01, sH1, HDIM, 0, 2 * w + 1, h_b1, lane, true);
      store_act(c10, sH1, HDIM, 1, 2 * w,     h_b1, lane, true);
      store_act(c11, sH1, HDIM, 1, 2 * w + 1, h_b1, lane, true);
    }
    __syncthreads();
    {                                            // hid2h: KT=8
      v8f c00 = {}, c01 = {}, c10 = {}, c11 = {};
      gemm_block22(sH1, HDIM, hW2, 8, 2 * w, lane, c00, c01, c10, c11);
      store_act(c00, sH2, HDIM, 0, 2 * w,     h_b2, lane, true);
      store_act(c01, sH2, HDIM, 0, 2 * w + 1, h_b2, lane, true);
      store_act(c10, sH2, HDIM, 1, 2 * w,     h_b2, lane, true);
      store_act(c11, sH2, HDIM, 1, 2 * w + 1, h_b2, lane, true);
    }
    __syncthreads();
    if (w == 0) {                                // y: N=16 -> 2x1 block
      v8f c0 = {}, c1 = {};
      gemm_block21(sX, 160, hrW, 4, 0, lane, c0, c1);
      gemm_block21(sH2, HDIM, hW3, 8, 0, lane, c0, c1);
      int n = lane & 15, h = lane >> 4;
      float bb = h_rb[n] + h_b3[n];
#pragma unroll
      for (int v = 0; v < 8; ++v) {
        out[((size_t)(row0 + v + h * 8) * T_LEN + t) * NY + n] = c0[v] + bb;
        out[((size_t)(row0 + 16 + v + h * 8) * T_LEN + t) * NY + n] =
            c1[v] + bb;
      }
    }

    // ---- f network: x_next = f([x|u]) ----   (sH1 free again; wave0's y
    // read sH2/sX only, so overlapping is safe)
    {                                            // hid1f: KT=5 (K=160)
      v8f c00 = {}, c01 = {}, c10 = {}, c11 = {};
      gemm_block22(sX, 160, fW1, 5, 2 * w, lane, c00, c01, c10, c11);
      store_act(c00, sH1, HDIM, 0, 2 * w,     f_b1, lane, true);
      store_act(c01, sH1, HDIM, 0, 2 * w + 1, f_b1, lane, true);
      store_act(c10, sH1, HDIM, 1, 2 * w,     f_b1, lane, true);
      store_act(c11, sH1, HDIM, 1, 2 * w + 1, f_b1, lane, true);
    }
    __syncthreads();
    {                                            // hid2f: KT=8
      v8f c00 = {}, c01 = {}, c10 = {}, c11 = {};
      gemm_block22(sH1, HDIM, fW2, 8, 2 * w, lane, c00, c01, c10, c11);
      store_act(c00, sH2, HDIM, 0, 2 * w,     f_b2, lane, true);
      store_act(c01, sH2, HDIM, 0, 2 * w + 1, f_b2, lane, true);
      store_act(c10, sH2, HDIM, 1, 2 * w,     f_b2, lane, true);
      store_act(c11, sH2, HDIM, 1, 2 * w + 1, f_b2, lane, true);
    }
    __syncthreads();
    // x_next: N=128, nt = w, 2x1 block into registers first
    v8f c0 = {}, c1 = {};
    gemm_block21(sX, 160, frW, 5, w, lane, c0, c1);
    gemm_block21(sH2, HDIM, fW3, 8, w, lane, c0, c1);
    __syncthreads();   // all skip-connection reads of sX complete
    {
      int n = lane & 15, h = lane >> 4;
      float bb = f_rb[w * 16 + n] + f_b3[w * 16 + n];
#pragma unroll
      for (int v = 0; v < 8; ++v) {
        sX[(v + h * 8) * 160 + w * 16 + n] = f2bf(c0[v] + bb);
        sX[(16 + v + h * 8) * 160 + w * 16 + n] = f2bf(c1[v] + bb);
      }
    }
    __syncthreads();
  }
}

// ---------------------------------------------------------------------------
extern "C" void kernel_launch(void* const* d_in, const int* in_sizes, int n_in,
                              void* d_out, int out_size, void* d_ws,
                              size_t ws_size, hipStream_t stream) {
  const float* upast   = (const float*)d_in[0];
  const float* ypast   = (const float*)d_in[1];
  const float* ufuture = (const float*)d_in[2];
  const float* enc_rW = (const float*)d_in[3];  const float* enc_rb = (const float*)d_in[4];
  const float* enc_W1 = (const float*)d_in[5];  const float* enc_b1 = (const float*)d_in[6];
  const float* enc_W2 = (const float*)d_in[7];  const float* enc_b2 = (const float*)d_in[8];
  const float* enc_W3 = (const float*)d_in[9];  const float* enc_b3 = (const float*)d_in[10];
  const float* f_rW = (const float*)d_in[11];   const float* f_rb = (const float*)d_in[12];
  const float* f_W1 = (const float*)d_in[13];   const float* f_b1 = (const float*)d_in[14];
  const float* f_W2 = (const float*)d_in[15];   const float* f_b2 = (const float*)d_in[16];
  const float* f_W3 = (const float*)d_in[17];   const float* f_b3 = (const float*)d_in[18];
  const float* h_rW = (const float*)d_in[19];   const float* h_rb = (const float*)d_in[20];
  const float* h_W1 = (const float*)d_in[21];   const float* h_b1 = (const float*)d_in[22];
  const float* h_W2 = (const float*)d_in[23];   const float* h_b2 = (const float*)d_in[24];
  const float* h_W3 = (const float*)d_in[25];   const float* h_b3 = (const float*)d_in[26];
  float* out = (float*)d_out;

  // Workspace layout (bf16 elements, 256B-aligned chunks). Total ~2.5 MB.
  __bf16* wsb = (__bf16*)d_ws;
  size_t off = 0;
  auto take = [&](size_t elems) {
    __bf16* p = wsb + off;
    off += (elems + 127) & ~(size_t)127;
    return p;
  };
  __bf16* x0s     = take((size_t)B_TOT * NX);
  __bf16* enc_rWs = take(1024 * 128);
  __bf16* enc_W1s = take(1024 * 256);
  __bf16* enc_W2s = take(256 * 256);
  __bf16* enc_W3s = take(256 * 128);
  __bf16* f_rWs   = take(160 * 128);
  __bf16* f_W1s   = take(160 * 256);
  __bf16* f_W2s   = take(256 * 256);
  __bf16* f_W3s   = take(256 * 128);
  __bf16* h_rWs   = take(128 * 16);
  __bf16* h_W1s   = take(128 * 256);
  __bf16* h_W2s   = take(256 * 256);
  __bf16* h_W3s   = take(256 * 16);

  auto swz = [&](const float* W, __bf16* dst, int Kreal, int Kpad, int N) {
    int total = Kpad * N;
    swizzle_weight_kernel<<<(total + 255) / 256, 256, 0, stream>>>(
        W, dst, Kreal, Kpad, N);
  };
  swz(enc_rW, enc_rWs, 1024, 1024, 128);
  swz(enc_W1, enc_W1s, 1024, 1024, 256);
  swz(enc_W2, enc_W2s, 256, 256, 256);
  swz(enc_W3, enc_W3s, 256, 256, 128);
  swz(f_rW, f_rWs, 144, 160, 128);
  swz(f_W1, f_W1s, 144, 160, 256);
  swz(f_W2, f_W2s, 256, 256, 256);
  swz(f_W3, f_W3s, 256, 256, 128);
  swz(h_rW, h_rWs, 128, 128, 16);
  swz(h_W1, h_W1s, 128, 128, 256);
  swz(h_W2, h_W2s, 256, 256, 256);
  swz(h_W3, h_W3s, 256, 256, 16);

  encoder_kernel<<<B_TOT / BM, 256, 0, stream>>>(
      upast, ypast, enc_rWs, enc_W1s, enc_W2s, enc_W3s,
      enc_rb, enc_b1, enc_b2, enc_b3, x0s);

  scan_kernel<<<B_TOT / BM, 256, 0, stream>>>(
      x0s, ufuture,
      f_rWs, f_W1s, f_W2s, f_W3s, f_rb, f_b1, f_b2, f_b3,
      h_rWs, h_W1s, h_W2s, h_W3s, h_rb, h_b1, h_b2, h_b3,
      out);

  (void)in_sizes; (void)n_in; (void)out_size; (void)ws_size;
}
